// Block_25409026523806
// MI455X (gfx1250) — matrix operand order repaired and checked by reading, not compiled, not run on hardware
//
#include <hip/hip_runtime.h>
#include <hip/hip_bf16.h>

// ---------------------------------------------------------------------------
// Types / helpers for CDNA5 (gfx1250, wave32) WMMA
// ---------------------------------------------------------------------------
typedef __bf16 bf16_t;
typedef __attribute__((ext_vector_type(16))) __bf16 v16bf;
typedef __attribute__((ext_vector_type(8)))  float  v8f;

__device__ __forceinline__ bf16_t f2bf(float f) {
    unsigned u = __builtin_bit_cast(unsigned, f);
    unsigned r = u + 0x7FFFu + ((u >> 16) & 1u);   // round-to-nearest-even
    unsigned short h = (unsigned short)(r >> 16);
    return __builtin_bit_cast(bf16_t, h);
}
__device__ __forceinline__ float bf2f(bf16_t b) {
    unsigned short h = __builtin_bit_cast(unsigned short, b);
    unsigned u = ((unsigned)h) << 16;
    return __builtin_bit_cast(float, u);
}

__device__ __forceinline__ v8f wmma_bf16(v16bf a, v16bf b, v8f c) {
    return __builtin_amdgcn_wmma_f32_16x16x32_bf16(
        /*neg_a=*/false, a, /*neg_b=*/false, b,
        /*c_mod=*/(short)0, c, /*reuse_a=*/false, /*reuse_b=*/false);
}

// Per-lane 16B async copy global -> LDS (whole wave moves 512B per issue).
__device__ __forceinline__ void async_cp_b128(uint32_t lds_off, const void* gsrc) {
    uint64_t g = (uint64_t)(uintptr_t)gsrc;
    asm volatile("global_load_async_to_lds_b128 %0, %1, off"
                 :: "v"(lds_off), "v"(g) : "memory");
}
__device__ __forceinline__ void wait_asynccnt0() {
#if __has_builtin(__builtin_amdgcn_s_wait_asynccnt)
    __builtin_amdgcn_s_wait_asynccnt(0);
#else
    asm volatile("s_wait_asynccnt 0" ::: "memory");
#endif
}

// A fragment: 16x32 bf16, row-major source, lane m = lane&15, kbase = (lane>>4)*16
__device__ __forceinline__ v16bf load_a(const bf16_t* p, int ld) {
    int lane = threadIdx.x & 31;
    int m = lane & 15, kb = (lane >> 4) * 16;
    const bf16_t* r = p + (size_t)m * ld;
    v16bf a;
#pragma unroll
    for (int i = 0; i < 16; ++i) a[i] = r[kb + i];
    return a;
}

// B fragment: 32x16 where B[k][n] = W[n][k], W row-major [N x K].
__device__ __forceinline__ v16bf load_b(const bf16_t* p, int ld) {
    int lane = threadIdx.x & 31;
    int n = lane & 15, kb = (lane >> 4) * 16;
    const bf16_t* r = p + (size_t)n * ld + kb;
    v16bf b;
#pragma unroll
    for (int i = 0; i < 16; ++i) b[i] = r[i];
    return b;
}

// B fragment from K-major storage: B[k][n] = V[k][n], V row-major [K x N].
__device__ __forceinline__ v16bf load_bk(const bf16_t* p, int ld) {
    int lane = threadIdx.x & 31;
    int n = lane & 15, kb = (lane >> 4) * 16;
    const bf16_t* r = p + (size_t)kb * ld + n;
    v16bf b;
#pragma unroll
    for (int i = 0; i < 16; ++i) b[i] = r[(size_t)i * ld];
    return b;
}

__device__ __forceinline__ void zero22(v8f c[2][2]) {
#pragma unroll
    for (int i = 0; i < 2; ++i)
#pragma unroll
        for (int j = 0; j < 2; ++j)
#pragma unroll
            for (int e = 0; e < 8; ++e) c[i][j][e] = 0.0f;
}

// ---------------------------------------------------------------------------
// GEMM core with async weight staging:
// block = 256 threads = 8 waves (2 m-waves x 4 n-waves), block tile 64x128.
// Per K-step: the block async-copies W[n0blk..+127][k0..+31] (8KB) into LDS
// (512 lane-chunks of 16B = 2 async issues per thread), waits ASYNCcnt,
// barriers, then every wave reads its B fragments from LDS while A comes
// straight from global (rows shared via L0/L2).
// ---------------------------------------------------------------------------
__device__ __forceinline__ void gemm_core_lds(const bf16_t* __restrict__ A,
                                              const bf16_t* __restrict__ W,
                                              int Kdim, int lda, int ldw,
                                              int m0, int n0blk, int wn,
                                              bf16_t* ldsW, v8f c[2][2]) {
    const uint32_t lbase = (uint32_t)(uintptr_t)ldsW;   // LDS aperture: [31:0] = offset
    const int tid = threadIdx.x;
    for (int k0 = 0; k0 < Kdim; k0 += 32) {
        // cooperative async stage: chunk cc -> row = cc>>2, 8-elem part = cc&3
#pragma unroll
        for (int t = 0; t < 2; ++t) {
            int cc = tid + t * 256;
            int row = cc >> 2, part = cc & 3;
            const bf16_t* src = W + (size_t)(n0blk + row) * ldw + k0 + part * 8;
            uint32_t dst = lbase + (uint32_t)(row * 32 + part * 8) * 2u;
            async_cp_b128(dst, src);
        }
        if (k0 + 32 < Kdim)
            __builtin_prefetch(A + (size_t)m0 * lda + k0 + 32, 0, 1);
        wait_asynccnt0();
        __syncthreads();

        v16bf a0 = load_a(A + (size_t)m0 * lda + k0, lda);
        v16bf a1 = load_a(A + (size_t)(m0 + 16) * lda + k0, lda);
        v16bf b0 = load_b(ldsW + (size_t)(wn * 32) * 32, 32);
        v16bf b1 = load_b(ldsW + (size_t)(wn * 32 + 16) * 32, 32);
        c[0][0] = wmma_bf16(a0, b0, c[0][0]);
        c[0][1] = wmma_bf16(a0, b1, c[0][1]);
        c[1][0] = wmma_bf16(a1, b0, c[1][0]);
        c[1][1] = wmma_bf16(a1, b1, c[1][1]);
        __syncthreads();   // protect LDS tile before next-iteration overwrite
    }
}

// ---------------------------------------------------------------------------
// Model constants
// ---------------------------------------------------------------------------
#define DIM 1024
#define NH 16
#define HD 64
#define ED 4096
#define BB 4
#define SS 1024
#define NTOK (BB * SS)   // 4096

// ---------------------------------------------------------------------------
// Elementwise / norm / gate kernels
// ---------------------------------------------------------------------------
__global__ void k_cvt(const float* __restrict__ in, bf16_t* __restrict__ out, int n) {
    int i = blockIdx.x * 256 + threadIdx.x;
    if (i < n) out[i] = f2bf(in[i]);
}

__global__ void k_rmsnorm(const float* __restrict__ x, const float* __restrict__ w,
                          bf16_t* __restrict__ out) {
    __shared__ float red[256];
    int t = blockIdx.x, tid = threadIdx.x;
    const float* xr = x + (size_t)t * DIM;
    float s = 0.f;
    for (int i = tid; i < DIM; i += 256) { float v = xr[i]; s += v * v; }
    red[tid] = s; __syncthreads();
    for (int o = 128; o > 0; o >>= 1) { if (tid < o) red[tid] += red[tid + o]; __syncthreads(); }
    float rinv = rsqrtf(red[0] * (1.0f / DIM) + 1e-6f);
    for (int i = tid; i < DIM; i += 256) out[(size_t)t * DIM + i] = f2bf(xr[i] * rinv * w[i]);
}

// gate: one block per token, one wave per expert (NE=8 waves = 256 threads)
__global__ void k_gate(const bf16_t* __restrict__ h2, const float* __restrict__ gw,
                       float* __restrict__ scale) {
    __shared__ float g[8];
    int t = blockIdx.x, tid = threadIdx.x;
    int wid = tid >> 5, lane = tid & 31;
    const bf16_t* xr = h2 + (size_t)t * DIM;
    const float* gr = gw + (size_t)wid * DIM;
    float s = 0.f;
    for (int i = lane; i < DIM; i += 32) s += bf2f(xr[i]) * gr[i];
#pragma unroll
    for (int off = 16; off > 0; off >>= 1) s += __shfl_xor(s, off);
    if (lane == 0) g[wid] = s;
    __syncthreads();
    if (tid == 0) {
        float mx = g[0];
        for (int i = 1; i < 8; ++i) mx = fmaxf(mx, g[i]);
        float e[8], sum = 0.f;
        for (int i = 0; i < 8; ++i) { e[i] = expf(g[i] - mx); sum += e[i]; }
        float m1 = -1.f, m2 = -1.f;
        for (int i = 0; i < 8; ++i) {
            if (e[i] > m1) { m2 = m1; m1 = e[i]; } else if (e[i] > m2) m2 = e[i];
        }
        float p2 = (m1 + m2) / sum;           // sum of top-2 softmax probs
        scale[t] = p2 / (p2 + 1e-9f);          // normalized top-2 weight sum
    }
}

// ---------------------------------------------------------------------------
// GEMM kernels
// ---------------------------------------------------------------------------
__global__ void __launch_bounds__(256) k_gemm_qkv(const bf16_t* __restrict__ A,
                                                  const bf16_t* __restrict__ W,
                                                  bf16_t* __restrict__ q,
                                                  bf16_t* __restrict__ k,
                                                  bf16_t* __restrict__ v) {
    __shared__ __align__(16) bf16_t ldsW[128 * 32];
    int wid = threadIdx.x >> 5, wm = wid & 1, wn = wid >> 1;
    int m0 = blockIdx.y * 64 + wm * 32;
    int n0blk = blockIdx.x * 128;
    int n0 = n0blk + wn * 32;
    v8f c[2][2]; zero22(c);
    gemm_core_lds(A, W, DIM, DIM, DIM, m0, n0blk, wn, ldsW, c);
    int lane = threadIdx.x & 31, n = lane & 15, hl = lane >> 4;
#pragma unroll
    for (int mi = 0; mi < 2; ++mi)
#pragma unroll
        for (int ni = 0; ni < 2; ++ni)
#pragma unroll
            for (int r = 0; r < 8; ++r) {
                int row = m0 + mi * 16 + r + 8 * hl;
                int col = n0 + ni * 16 + n;
                int which = col >> 10, rem = col & 1023;
                int h = rem >> 6, dd = rem & 63;
                int b_ = row >> 10, s_ = row & 1023;
                size_t dst = (((size_t)(b_ * NH + h) * SS + s_) * HD + dd);
                bf16_t val = f2bf(c[mi][ni][r]);
                if (which == 0) q[dst] = val;
                else if (which == 1) k[dst] = val;
                else v[dst] = val;
            }
}

__global__ void __launch_bounds__(256) k_gemm_o(const bf16_t* __restrict__ A,
                                                const bf16_t* __restrict__ W,
                                                const float* __restrict__ x,
                                                float* __restrict__ x2) {
    __shared__ __align__(16) bf16_t ldsW[128 * 32];
    int wid = threadIdx.x >> 5, wm = wid & 1, wn = wid >> 1;
    int m0 = blockIdx.y * 64 + wm * 32;
    int n0blk = blockIdx.x * 128;
    int n0 = n0blk + wn * 32;
    v8f c[2][2]; zero22(c);
    gemm_core_lds(A, W, DIM, DIM, DIM, m0, n0blk, wn, ldsW, c);
    int lane = threadIdx.x & 31, n = lane & 15, hl = lane >> 4;
#pragma unroll
    for (int mi = 0; mi < 2; ++mi)
#pragma unroll
        for (int ni = 0; ni < 2; ++ni)
#pragma unroll
            for (int r = 0; r < 8; ++r) {
                size_t row = m0 + mi * 16 + r + 8 * hl;
                size_t col = n0 + ni * 16 + n;
                size_t idx = row * DIM + col;
                x2[idx] = x[idx] + c[mi][ni][r];
            }
}

__global__ void __launch_bounds__(256) k_gemm_up(const bf16_t* __restrict__ A,
                                                 const bf16_t* __restrict__ W,
                                                 bf16_t* __restrict__ act) {
    __shared__ __align__(16) bf16_t ldsW[128 * 32];
    int wid = threadIdx.x >> 5, wm = wid & 1, wn = wid >> 1;
    int m0 = blockIdx.y * 64 + wm * 32;
    int n0blk = blockIdx.x * 128;
    int n0 = n0blk + wn * 32;
    v8f c[2][2]; zero22(c);
    gemm_core_lds(A, W, DIM, DIM, DIM, m0, n0blk, wn, ldsW, c);
    int lane = threadIdx.x & 31, n = lane & 15, hl = lane >> 4;
#pragma unroll
    for (int mi = 0; mi < 2; ++mi)
#pragma unroll
        for (int ni = 0; ni < 2; ++ni)
#pragma unroll
            for (int r = 0; r < 8; ++r) {
                size_t row = m0 + mi * 16 + r + 8 * hl;
                size_t col = n0 + ni * 16 + n;
                float u = c[mi][ni][r];
                float su = u / (1.0f + __expf(-u));   // silu
                act[row * ED + col] = f2bf(su);
            }
}

__global__ void __launch_bounds__(256) k_gemm_down(const bf16_t* __restrict__ A,
                                                   const bf16_t* __restrict__ W,
                                                   const float* __restrict__ x2,
                                                   const float* __restrict__ scale,
                                                   float* __restrict__ out) {
    __shared__ __align__(16) bf16_t ldsW[128 * 32];
    int wid = threadIdx.x >> 5, wm = wid & 1, wn = wid >> 1;
    int m0 = blockIdx.y * 64 + wm * 32;
    int n0blk = blockIdx.x * 128;
    int n0 = n0blk + wn * 32;
    v8f c[2][2]; zero22(c);
    gemm_core_lds(A, W, ED, ED, ED, m0, n0blk, wn, ldsW, c);
    int lane = threadIdx.x & 31, n = lane & 15, hl = lane >> 4;
#pragma unroll
    for (int mi = 0; mi < 2; ++mi)
#pragma unroll
        for (int ni = 0; ni < 2; ++ni)
#pragma unroll
            for (int r = 0; r < 8; ++r) {
                size_t row = m0 + mi * 16 + r + 8 * hl;
                size_t col = n0 + ni * 16 + n;
                size_t idx = row * DIM + col;
                out[idx] = x2[idx] + c[mi][ni][r] * scale[row];
            }
}

// ---------------------------------------------------------------------------
// Flash attention: 1 wave = 16 q-rows, k-tiles of 32, online softmax, P via LDS
// block = 128 threads (4 waves), grid = B*H*(S/16)/4
// ---------------------------------------------------------------------------
__global__ void __launch_bounds__(128) k_attn(const bf16_t* __restrict__ q,
                                              const bf16_t* __restrict__ k,
                                              const bf16_t* __restrict__ v,
                                              bf16_t* __restrict__ o) {
    __shared__ bf16_t psh[4 * 16 * 32];
    int wid = threadIdx.x >> 5, lane = threadIdx.x & 31;
    int g = blockIdx.x * 4 + wid;
    const int nqt = SS / 16;
    int qt = g % nqt;
    int bh = g / nqt;              // b*NH + h
    int h = bh % NH, b = bh / NH;
    const bf16_t* qp = q + (size_t)bh * SS * HD;
    const bf16_t* kp = k + (size_t)bh * SS * HD;
    const bf16_t* vp = v + (size_t)bh * SS * HD;
    int qbase = qt * 16;
    int n = lane & 15, hl = lane >> 4;

    v16bf qa0 = load_a(qp + (size_t)qbase * HD, HD);
    v16bf qa1 = load_a(qp + (size_t)qbase * HD + 32, HD);

    v8f o4[4];
#pragma unroll
    for (int nd = 0; nd < 4; ++nd)
#pragma unroll
        for (int e = 0; e < 8; ++e) o4[nd][e] = 0.f;
    float m8[8], l8[8];
#pragma unroll
    for (int r = 0; r < 8; ++r) { m8[r] = -3e38f; l8[r] = 0.f; }

    bf16_t* pl = psh + wid * 512;

    for (int k0 = 0; k0 <= qbase + 15; k0 += 32) {
        // S = Q K^T over this 16x32 tile of scores
        v8f s[2];
#pragma unroll
        for (int nf = 0; nf < 2; ++nf) {
#pragma unroll
            for (int e = 0; e < 8; ++e) s[nf][e] = 0.f;
            v16bf kb0 = load_b(kp + (size_t)(k0 + nf * 16) * HD, HD);
            v16bf kb1 = load_b(kp + (size_t)(k0 + nf * 16) * HD + 32, HD);
            s[nf] = wmma_bf16(qa0, kb0, s[nf]);
            s[nf] = wmma_bf16(qa1, kb1, s[nf]);
        }
        // online softmax (row reductions across 16-lane halves)
        float alpha[8], pv0[8], pv1[8];
#pragma unroll
        for (int r = 0; r < 8; ++r) {
            int row = qbase + r + 8 * hl;
            float s0 = s[0][r] * 0.125f;          // hd^-0.5
            float s1 = s[1][r] * 0.125f;
            if (k0 + n > row) s0 = -3e38f;        // causal mask
            if (k0 + 16 + n > row) s1 = -3e38f;
            float mx = fmaxf(s0, s1);
#pragma unroll
            for (int off = 1; off < 16; off <<= 1) mx = fmaxf(mx, __shfl_xor(mx, off));
            float mnew = fmaxf(m8[r], mx);
            float a = __expf(m8[r] - mnew);
            float p0 = __expf(s0 - mnew), p1 = __expf(s1 - mnew);
            float rs = p0 + p1;
#pragma unroll
            for (int off = 1; off < 16; off <<= 1) rs += __shfl_xor(rs, off);
            l8[r] = l8[r] * a + rs;
            m8[r] = mnew; alpha[r] = a; pv0[r] = p0; pv1[r] = p1;
        }
#pragma unroll
        for (int nd = 0; nd < 4; ++nd)
#pragma unroll
            for (int r = 0; r < 8; ++r) o4[nd][r] *= alpha[r];
        // P: C-layout -> LDS -> A-layout (wave-private region, no barrier needed)
#pragma unroll
        for (int r = 0; r < 8; ++r) {
            pl[(size_t)(r + 8 * hl) * 32 + n]      = f2bf(pv0[r]);
            pl[(size_t)(r + 8 * hl) * 32 + 16 + n] = f2bf(pv1[r]);
        }
        asm volatile("s_wait_dscnt 0" ::: "memory");
        v16bf pa = load_a(pl, 32);
        // O += P V
#pragma unroll
        for (int nd = 0; nd < 4; ++nd) {
            v16bf vb = load_bk(vp + (size_t)k0 * HD + nd * 16, HD);
            o4[nd] = wmma_bf16(pa, vb, o4[nd]);
        }
    }
    // epilogue: normalize and scatter to [B,S,D] with column h*64+d
#pragma unroll
    for (int nd = 0; nd < 4; ++nd)
#pragma unroll
        for (int r = 0; r < 8; ++r) {
            int row = qbase + r + 8 * hl;
            float val = o4[nd][r] / l8[r];
            size_t dst = ((size_t)(b * SS + row) * DIM) + h * HD + nd * 16 + n;
            o[dst] = f2bf(val);
        }
}

// ---------------------------------------------------------------------------
// Host launcher
// ---------------------------------------------------------------------------
extern "C" void kernel_launch(void* const* d_in, const int* in_sizes, int n_in,
                              void* d_out, int out_size, void* d_ws, size_t ws_size,
                              hipStream_t stream) {
    const float* x      = (const float*)d_in[0];
    const float* n1_w   = (const float*)d_in[1];
    const float* qkv_w  = (const float*)d_in[2];
    const float* o_w    = (const float*)d_in[3];
    const float* n2_w   = (const float*)d_in[4];
    const float* gate_w = (const float*)d_in[5];
    const float* up_w   = (const float*)d_in[6];
    const float* down_w = (const float*)d_in[7];
    float* out = (float*)d_out;

    // workspace layout (bytes)
    uint8_t* w = (uint8_t*)d_ws;
    size_t off = 0;
    auto take = [&](size_t bytes) { void* p = w + off; off += (bytes + 255) & ~size_t(255); return p; };
    bf16_t* h1    = (bf16_t*)take((size_t)NTOK * DIM * 2);
    bf16_t* qkvwb = (bf16_t*)take((size_t)3 * DIM * DIM * 2);
    bf16_t* owb   = (bf16_t*)take((size_t)DIM * DIM * 2);
    bf16_t* upwb  = (bf16_t*)take((size_t)ED * DIM * 2);
    bf16_t* dnwb  = (bf16_t*)take((size_t)DIM * ED * 2);
    bf16_t* qb    = (bf16_t*)take((size_t)NTOK * DIM * 2);
    bf16_t* kb    = (bf16_t*)take((size_t)NTOK * DIM * 2);
    bf16_t* vb    = (bf16_t*)take((size_t)NTOK * DIM * 2);
    bf16_t* ao    = (bf16_t*)take((size_t)NTOK * DIM * 2);
    float*  x2    = (float*) take((size_t)NTOK * DIM * 4);
    bf16_t* h2    = (bf16_t*)take((size_t)NTOK * DIM * 2);
    float*  scl   = (float*) take((size_t)NTOK * 4);
    bf16_t* act   = (bf16_t*)take((size_t)NTOK * ED * 2);
    (void)ws_size; (void)in_sizes; (void)n_in; (void)out_size;

    // 1. weights -> bf16
    k_cvt<<<(3 * DIM * DIM + 255) / 256, 256, 0, stream>>>(qkv_w, qkvwb, 3 * DIM * DIM);
    k_cvt<<<(DIM * DIM + 255) / 256, 256, 0, stream>>>(o_w, owb, DIM * DIM);
    k_cvt<<<(ED * DIM + 255) / 256, 256, 0, stream>>>(up_w, upwb, ED * DIM);
    k_cvt<<<(DIM * ED + 255) / 256, 256, 0, stream>>>(down_w, dnwb, DIM * ED);

    // 2. h1 = rmsnorm(x, n1_w)
    k_rmsnorm<<<NTOK, 256, 0, stream>>>(x, n1_w, h1);

    // 3. qkv = h1 @ qkv_w^T  -> q/k/v in [B,H,S,hd]
    k_gemm_qkv<<<dim3(3 * DIM / 128, NTOK / 64), 256, 0, stream>>>(h1, qkvwb, qb, kb, vb);

    // 4. flash attention -> ao [B,S,D]
    k_attn<<<(BB * NH * (SS / 16)) / 4, 128, 0, stream>>>(qb, kb, vb, ao);

    // 5. x2 = x + ao @ o_w^T
    k_gemm_o<<<dim3(DIM / 128, NTOK / 64), 256, 0, stream>>>(ao, owb, x, x2);

    // 6. h2 = rmsnorm(x2, n2_w)
    k_rmsnorm<<<NTOK, 256, 0, stream>>>(x2, n2_w, h2);

    // 7. per-token MoE gate scalar
    k_gate<<<NTOK, 256, 0, stream>>>(h2, gate_w, scl);

    // 8. act = silu(h2 @ up_w^T)
    k_gemm_up<<<dim3(ED / 128, NTOK / 64), 256, 0, stream>>>(h2, upwb, act);

    // 9. out = x2 + (act @ down_w^T) * scl
    k_gemm_down<<<dim3(DIM / 128, NTOK / 64), 256, 0, stream>>>(act, dnwb, x2, scl, out);
}
